// ChannelSelfCorrelation_82291573391859
// MI455X (gfx1250) — matrix operand order repaired and checked by reading, not compiled, not run on hardware
//
#include <hip/hip_runtime.h>

typedef unsigned short u16;
typedef __attribute__((ext_vector_type(4)))  u16    v4u;
typedef __attribute__((ext_vector_type(8)))  u16    v8u;
typedef __attribute__((ext_vector_type(16))) u16    v16u;
typedef __attribute__((ext_vector_type(16))) __bf16 v16bf;
typedef __attribute__((ext_vector_type(8)))  float  v8f;

#define LDIM 1024

// ---------------- helpers ----------------

__device__ __forceinline__ u16 f2bf(float f) {
  unsigned u = __builtin_bit_cast(unsigned, f);
  u += 0x7FFFu + ((u >> 16) & 1u);        // round-to-nearest-even
  return (u16)(u >> 16);
}

// A fragment (16x32 bf16, M x K), source row-major (lane&15 = M row).
// Per ISA: lanes 0-15 hold K = {k0+0..7, k0+16..23}; lanes 16-31 hold {k0+8..15, k0+24..31}.
__device__ __forceinline__ v16bf load_frag_a(const u16* __restrict__ p, int row, int k0, int lane) {
  const u16* q = p + (size_t)(row + (lane & 15)) * LDIM + k0 + ((lane >> 4) << 3);
  v8u lo = *(const v8u*)q;
  v8u hi = *(const v8u*)(q + 16);
  v16u f = __builtin_shufflevector(lo, hi, 0,1,2,3,4,5,6,7,8,9,10,11,12,13,14,15);
  return __builtin_bit_cast(v16bf, f);
}

// B fragment (32x16 bf16, K x N), sourced from the TRANSPOSE of B stored row-major
// (so T[n,k] = B[k,n]; lane&15 = N column;
//  lanes 0-15 -> K = {k0+0..7, k0+16..23}, lanes 16-31 -> K = {k0+8..15, k0+24..31}).
__device__ __forceinline__ v16bf load_frag_b(const u16* __restrict__ p, int col, int k0, int lane) {
  const u16* q = p + (size_t)(col + (lane & 15)) * LDIM + k0 + ((lane >> 4) << 3);
  v8u lo = *(const v8u*)q;
  v8u hi = *(const v8u*)(q + 16);
  v16u f = __builtin_shufflevector(lo, hi, 0,1,2,3,4,5,6,7,8,9,10,11,12,13,14,15);
  return __builtin_bit_cast(v16bf, f);
}

// ---------------- small utility kernels ----------------

__global__ __launch_bounds__(256) void cvt_f32_bf16_k(const float* __restrict__ src,
                                                      u16* __restrict__ dst) {
  size_t i = ((size_t)blockIdx.x * 256 + threadIdx.x) * 4;
  float4 v = *(const float4*)(src + i);
  v4u pk;
  pk[0] = f2bf(v.x); pk[1] = f2bf(v.y); pk[2] = f2bf(v.z); pk[3] = f2bf(v.w);
  *(v4u*)(dst + i) = pk;
}

// Xt[b][n][c] = x[b][c][n], f32 -> bf16, 32x32 LDS-tiled transpose.
__global__ __launch_bounds__(256) void transpose_bf16_k(const float* __restrict__ x,
                                                        u16* __restrict__ xt) {
  __shared__ float tile[32][33];
  const int b  = blockIdx.z;
  const float* X = x  + ((size_t)b << 20);
  u16*         T = xt + ((size_t)b << 20);
  const int c0 = blockIdx.y * 32, n0 = blockIdx.x * 32;
  const int tx = threadIdx.x, ty = threadIdx.y;
#pragma unroll
  for (int i = 0; i < 4; ++i)
    tile[ty + 8 * i][tx] = X[(size_t)(c0 + ty + 8 * i) * LDIM + n0 + tx];
  __syncthreads();
#pragma unroll
  for (int i = 0; i < 4; ++i)
    T[(size_t)(n0 + ty + 8 * i) * LDIM + c0 + tx] = f2bf(tile[tx][ty + 8 * i]);
}

// Row softmax over 1024 f32, output bf16. One 256-thread block per row.
__global__ __launch_bounds__(256) void softmax_k(const float* __restrict__ S,
                                                 u16* __restrict__ A) {
  const int row = blockIdx.x;
  const int tid = threadIdx.x;
  float4 v = ((const float4*)(S + (size_t)row * LDIM))[tid];
  __shared__ float red[256];
  float mx = fmaxf(fmaxf(v.x, v.y), fmaxf(v.z, v.w));
  red[tid] = mx; __syncthreads();
  for (int s = 128; s > 0; s >>= 1) {
    if (tid < s) red[tid] = fmaxf(red[tid], red[tid + s]);
    __syncthreads();
  }
  mx = red[0]; __syncthreads();
  float e0 = __expf(v.x - mx), e1 = __expf(v.y - mx);
  float e2 = __expf(v.z - mx), e3 = __expf(v.w - mx);
  red[tid] = e0 + e1 + e2 + e3; __syncthreads();
  for (int s = 128; s > 0; s >>= 1) {
    if (tid < s) red[tid] += red[tid + s];
    __syncthreads();
  }
  float inv = 1.0f / red[0];
  v4u pk;
  pk[0] = f2bf(e0 * inv); pk[1] = f2bf(e1 * inv);
  pk[2] = f2bf(e2 * inv); pk[3] = f2bf(e3 * inv);
  *(v4u*)(A + (size_t)row * LDIM + tid * 4) = pk;
}

// ---------------- WMMA GEMM ----------------
// D(M,N) = A(M,K) * B(K,N), all dims 1024, per-batch via blockIdx.z.
// Bbase points at B-transposed row-major (T[n][k]).
// OUT_BF16_T: 1 -> store D transposed as bf16 (O[n*1024+m], contiguous v8u per lane)
//             0 -> store D row-major f32.
// BIAS: 0 none, 1 bias[n] (with transposed bf16 store), 2 bias[m] (with f32 store).
template <bool A_SHARED, bool B_SHARED, int OUT_BF16_T, int BIAS>
__global__ __launch_bounds__(128) void gemm_wmma_k(const u16* __restrict__ Abase,
                                                   const u16* __restrict__ Bbase,
                                                   const float* __restrict__ bias,
                                                   void* __restrict__ Obase) {
  const int lane  = threadIdx.x & 31;
  const int wave  = threadIdx.x >> 5;
  const int b     = blockIdx.z;
  const int waveN = blockIdx.x * 256 + wave * 64;
  const int waveM = blockIdx.y * 64;
  const u16* A  = Abase + (A_SHARED ? 0 : ((size_t)b << 20));
  const u16* Bt = Bbase + (B_SHARED ? 0 : ((size_t)b << 20));

  v8f acc[4][4] = {};
  for (int k0 = 0; k0 < 1024; k0 += 32) {
    v16bf af[4], bf[4];
#pragma unroll
    for (int i = 0; i < 4; ++i) af[i] = load_frag_a(A, waveM + 16 * i, k0, lane);
#pragma unroll
    for (int j = 0; j < 4; ++j) bf[j] = load_frag_b(Bt, waveN + 16 * j, k0, lane);
#pragma unroll
    for (int i = 0; i < 4; ++i)
#pragma unroll
      for (int j = 0; j < 4; ++j)
        acc[i][j] = __builtin_amdgcn_wmma_f32_16x16x32_bf16(
            false, af[i], false, bf[j], (short)0, acc[i][j], false, false);
  }

  if (OUT_BF16_T) {
    u16* O = (u16*)Obase + ((size_t)b << 20);
#pragma unroll
    for (int i = 0; i < 4; ++i)
#pragma unroll
      for (int j = 0; j < 4; ++j) {
        const int n  = waveN + 16 * j + (lane & 15);
        const int m0 = waveM + 16 * i + ((lane >> 4) << 3);
        const float bia = (BIAS == 1) ? bias[n] : 0.0f;
        v8u pk;
#pragma unroll
        for (int t = 0; t < 8; ++t) pk[t] = f2bf(acc[i][j][t] + bia);
        *(v8u*)(O + (size_t)n * LDIM + m0) = pk;
      }
  } else {
    float* O = (float*)Obase + ((size_t)b << 20);
#pragma unroll
    for (int i = 0; i < 4; ++i)
#pragma unroll
      for (int j = 0; j < 4; ++j) {
        const int n  = waveN + 16 * j + (lane & 15);
        const int m0 = waveM + 16 * i + ((lane >> 4) << 3);
#pragma unroll
        for (int t = 0; t < 8; ++t) {
          const int m = m0 + t;
          const float bia = (BIAS == 2) ? bias[m] : 0.0f;
          O[(size_t)m * LDIM + n] = acc[i][j][t] + bia;
        }
      }
  }
}

// ---------------- launch ----------------

extern "C" void kernel_launch(void* const* d_in, const int* in_sizes, int n_in,
                              void* d_out, int out_size, void* d_ws, size_t ws_size,
                              hipStream_t stream) {
  (void)in_sizes; (void)n_in; (void)out_size; (void)ws_size;
  const float* x  = (const float*)d_in[0];
  const float* Wq = (const float*)d_in[1];
  const float* bq = (const float*)d_in[2];
  const float* Wk = (const float*)d_in[3];
  const float* bk = (const float*)d_in[4];
  const float* Wo = (const float*)d_in[5];
  const float* bo = (const float*)d_in[6];
  float* Y = (float*)d_out;

  char* ws = (char*)d_ws;
  const size_t MB = 1u << 20;
  u16*   Xt  = (u16*)(ws);              // 64 MB  : Xt[b] = x[b]^T, bf16
  u16*   Wqb = (u16*)(ws + 64 * MB);    //  2 MB
  u16*   Wkb = (u16*)(ws + 66 * MB);    //  2 MB
  u16*   Wob = (u16*)(ws + 68 * MB);    //  2 MB
  u16*   Q   = (u16*)(ws + 70 * MB);    // 64 MB  : Q[b](ch,pix) bf16
  u16*   Kb  = (u16*)(ws + 134 * MB);   // 64 MB  : K[b](ch,pix) bf16
  float* S   = (float*)(ws + 198 * MB); // 128 MB : logits f32
  u16*   At  = Q;                       // alias: attn bf16 (Q dead after S-GEMM)
  u16*   Ot  = Kb;                      // alias: O^T bf16 (K dead after S-GEMM)

  // 1) precision casts + transpose
  cvt_f32_bf16_k<<<1024, 256, 0, stream>>>(Wq, Wqb);
  cvt_f32_bf16_k<<<1024, 256, 0, stream>>>(Wk, Wkb);
  cvt_f32_bf16_k<<<1024, 256, 0, stream>>>(Wo, Wob);
  transpose_bf16_k<<<dim3(32, 32, 32), dim3(32, 8), 0, stream>>>(x, Xt);

  const dim3 g(4, 16, 32), blk(128);
  // 2) Q = Wq*X + bq, K = Wk*X + bk  (computed as Xt*Wq^T with transposed epilogue)
  gemm_wmma_k<false, true, 1, 1><<<g, blk, 0, stream>>>(Xt, Wqb, bq, Q);
  gemm_wmma_k<false, true, 1, 1><<<g, blk, 0, stream>>>(Xt, Wkb, bk, Kb);
  // 3) S = Q * K^T  (f32 logits)
  gemm_wmma_k<false, false, 0, 0><<<g, blk, 0, stream>>>(Q, Kb, nullptr, S);
  // 4) row softmax -> bf16 attn
  softmax_k<<<32768, 256, 0, stream>>>(S, At);
  // 5) O = A * X  (store O^T bf16)
  gemm_wmma_k<false, false, 1, 0><<<g, blk, 0, stream>>>(At, Xt, nullptr, Ot);
  // 6) Y = Wo * O + bo  (f32 out)
  gemm_wmma_k<true, false, 0, 2><<<g, blk, 0, stream>>>(Wob, Ot, bo, Y);
}
